// interRAL_70626442215557
// MI455X (gfx1250) — matrix operand order repaired and checked by reading, not compiled, not run on hardware
//
#include <hip/hip_runtime.h>
#include <hip/hip_bf16.h>

typedef __attribute__((ext_vector_type(16))) _Float16 v16h;
typedef __attribute__((ext_vector_type(8)))  _Float16 v8h;
typedef __attribute__((ext_vector_type(8)))  float    v8f;
typedef __attribute__((ext_vector_type(4)))  float    v4f;

#define B_DIM 64
#define N_PTS 8192
#define C_DIM 64
#define NPW   16                  // points per workgroup in pass 1
#define PASS1_WGS (N_PTS / NPW)   // 512

// ---------------- WMMA helpers (CDNA5 wave32, 16x16x32 f16 -> f32) ----------------

__device__ __forceinline__ v8f wmma_f16(v16h a, v16h b, v8f c) {
  // 8 args: (neg_a, A, neg_b, B, c_mod, C, reuse_a, reuse_b)
  return __builtin_amdgcn_wmma_f32_16x16x32_f16(false, a, false, b, (short)0, c, false, false);
}

#define CAT16 0,1,2,3,4,5,6,7,8,9,10,11,12,13,14,15

// A fragment, row-major f16 [M][64], rows m0..+15, K window [k0,k0+32).
// ISA: lanes 0-15 hold K={0..7,16..23}, lanes 16-31 hold K={8..15,24..31}; row = lane%16.
__device__ __forceinline__ v16h load_a_rm64(const _Float16* src, int m0, int k0) {
  int lane = threadIdx.x & 31;
  int m  = m0 + (lane & 15);
  int kb = k0 + ((lane >> 4) << 3);
  v8h lo = *(const v8h*)(src + m * 64 + kb);
  v8h hi = *(const v8h*)(src + m * 64 + kb + 16);
  return __builtin_shufflevector(lo, hi, CAT16);
}

// A fragment from f16 [M][16]: K valid only 0..15, upper 16 K zero-padded.
__device__ __forceinline__ v16h load_a_k16(const _Float16* src, int m0) {
  int lane = threadIdx.x & 31;
  int m  = m0 + (lane & 15);
  int kb = (lane >> 4) << 3;
  v8h lo = *(const v8h*)(src + m * 16 + kb);
  v8h z  = {};
  return __builtin_shufflevector(lo, z, CAT16);
}

// B fragment where B[k][n] = src[n*64 + k]  (row-major [N][K] storage: W^T / X / attn etc).
// ISA: lanes 0-15 hold K={0..7,16..23}, lanes 16-31 hold K={8..15,24..31}; col = lane%16.
__device__ __forceinline__ v16h load_b_nk64(const _Float16* src, int k0, int n0) {
  int lane = threadIdx.x & 31;
  int n  = n0 + (lane & 15);
  int kb = k0 + ((lane >> 4) << 3);
  v8h lo = *(const v8h*)(src + n * 64 + kb);
  v8h hi = *(const v8h*)(src + n * 64 + kb + 16);
  return __builtin_shufflevector(lo, hi, CAT16);
}

// B fragment from f16 [N][16], K valid 0..15 (upper 16 K zero-padded).
__device__ __forceinline__ v16h load_b_k16(const _Float16* src, int n0) {
  int lane = threadIdx.x & 31;
  int n  = n0 + (lane & 15);
  v8h lo = *(const v8h*)(src + n * 16 + ((lane >> 4) << 3));
  v8h z  = {};
  return __builtin_shufflevector(lo, z, CAT16);
}

// ---------------- Pass 1: fused per-point attention -> t[N][B][C] + BN partial sums ----
//
// All products whose results must land row-major in LDS are computed TRANSPOSED
// (M^T = B^T A^T), so each lane's 8 D-fragment values are contiguous -> 1x ds_store_b128.

// LDS map (bytes), total 61696 < 64KB. E(f32 64x64) overlaps {Yh, xrawh, yrawh} early, attnh late.
#define SMEM_BYTES 61696
#define OFF_E      0        // f32 [64][64]  (16384)
#define OFF_YH     0        // f16 [64][64]  (8192)   dead before energy written
#define OFF_XRAWH  8192     // f16 [64][16]  (2048)   x row, K zero-padded to 16
#define OFF_YRAWH  10240    // f16 [64][16]  (2048)
#define OFF_ATTNH  0        // f16 [64][64]  (8192)   in-place f32->f16 after barrier
#define OFF_XH     16384    // f16 [64][64]  (8192)   X, overwritten in-place by Z=X-yr
#define OFF_WQK    24576    // f16 [16][64]  (2048)   [d][c]
#define OFF_WV     26624    // f16 [64][64]  (8192)   [d][c]
#define OFF_WT     34816    // f16 [64][64]  (8192)   [d][c]
#define OFF_WLT    43008    // f16 [64][16]  (2048)   Wl TRANSPOSED [c][k], zero-padded
#define OFF_WRT    45056    // f16 [64][16]  (2048)   Wr TRANSPOSED [c][k], zero-padded
#define OFF_BL     47104    // f32 [64]
#define OFF_BR     47360
#define OFF_BV     47616
#define OFF_BT     47872
#define OFF_XQH    48128    // f16 [64][16]  (2048)   xq row-major [b][d]
#define OFF_YKH    50176    // f16 [64][16]  (2048)   yk row-major [e][d]
#define OFF_YVT    52224    // f16 [64][64]  (8192)   yv TRANSPOSED [c][e]
#define OFF_COLSUM 60416    // f32 [64]
#define OFF_RED    60672    // f32 [4][64]   (1024)

__global__ __launch_bounds__(256) void attn_pass1(
    const float* __restrict__ x,  const float* __restrict__ y,
    const float* __restrict__ W_l, const float* __restrict__ b_l,
    const float* __restrict__ W_r, const float* __restrict__ b_r,
    const float* __restrict__ W_qk, const float* __restrict__ W_v, const float* __restrict__ b_v,
    const float* __restrict__ W_t,  const float* __restrict__ b_t,
    float* __restrict__ t_out, float* __restrict__ bn_sums)
{
  __shared__ __align__(16) unsigned char smem[SMEM_BYTES];
  float*     E      = (float*)(smem + OFF_E);
  _Float16*  Yh     = (_Float16*)(smem + OFF_YH);
  _Float16*  xrawh  = (_Float16*)(smem + OFF_XRAWH);
  _Float16*  yrawh  = (_Float16*)(smem + OFF_YRAWH);
  _Float16*  attnh  = (_Float16*)(smem + OFF_ATTNH);
  _Float16*  Xh     = (_Float16*)(smem + OFF_XH);     // Z after phase 7
  _Float16*  Wqkh   = (_Float16*)(smem + OFF_WQK);
  _Float16*  Wvh    = (_Float16*)(smem + OFF_WV);
  _Float16*  Wth    = (_Float16*)(smem + OFF_WT);
  _Float16*  WlT    = (_Float16*)(smem + OFF_WLT);
  _Float16*  WrT    = (_Float16*)(smem + OFF_WRT);
  float*     bl_s   = (float*)(smem + OFF_BL);
  float*     br_s   = (float*)(smem + OFF_BR);
  float*     bv_s   = (float*)(smem + OFF_BV);
  float*     bt_s   = (float*)(smem + OFF_BT);
  _Float16*  xqh    = (_Float16*)(smem + OFF_XQH);
  _Float16*  ykh    = (_Float16*)(smem + OFF_YKH);
  _Float16*  yvT    = (_Float16*)(smem + OFF_YVT);
  float*     colsum = (float*)(smem + OFF_COLSUM);
  float*     red_s  = (float*)(smem + OFF_RED);

  const int tid  = threadIdx.x;
  const int lane = tid & 31;
  const int wave = tid >> 5;                       // 0..7
  const int t0 = wave * 2, t1 = wave * 2 + 1;      // 2 tiles of the 4x4 tiling per wave
  const int m0_0 = (t0 >> 2) * 16, n0_0 = (t0 & 3) * 16;
  const int m0_1 = (t1 >> 2) * 16, n0_1 = (t1 & 3) * 16;   // note: m0_0 == m0_1 always
  const int hi = (lane >> 4) & 1;
  const int nn = lane & 15;

  // ---- cache weights once per workgroup (f16) ----
  for (int i = tid; i < 16 * 64; i += 256) Wqkh[i] = (_Float16)W_qk[i];
  for (int i = tid; i < 64 * 64; i += 256) Wvh[i]  = (_Float16)W_v[i];
  for (int i = tid; i < 64 * 64; i += 256) Wth[i]  = (_Float16)W_t[i];
  for (int i = tid; i < 1024; i += 256) {          // transposed, K zero-padded input weights
    int c = i >> 4, k = i & 15;
    WlT[i] = (k < 10) ? (_Float16)W_l[k * 64 + c] : (_Float16)0.f;
    WrT[i] = (k < 13) ? (_Float16)W_r[k * 64 + c] : (_Float16)0.f;
  }
  if (tid < 64) { bl_s[tid] = b_l[tid]; br_s[tid] = b_r[tid]; bv_s[tid] = b_v[tid]; bt_s[tid] = b_t[tid]; }
  __syncthreads();

  v8f bnS = {}, bnQ = {};   // BN partials; both wave tiles share the same channel range

  for (int ni = 0; ni < NPW; ++ni) {
    const int n = blockIdx.x * NPW + ni;

    // phase 0: gather x[:,n,:], y[:,n,:] as zero-padded f16 tiles; prefetch 4 points ahead
    for (int i = tid; i < 1024; i += 256) {
      int b = i >> 4, k = i & 15;
      xrawh[i] = (k < 10) ? (_Float16)x[((size_t)b * N_PTS + n) * 10 + k] : (_Float16)0.f;
      yrawh[i] = (k < 13) ? (_Float16)y[((size_t)b * N_PTS + n) * 13 + k] : (_Float16)0.f;
    }
    if (ni + 4 < NPW && tid < 64) {
      __builtin_prefetch(&x[((size_t)tid * N_PTS + (n + 4)) * 10], 0, 1);
      __builtin_prefetch(&y[((size_t)tid * N_PTS + (n + 4)) * 13], 0, 1);
    }
    __syncthreads();

    // phase 1 (WMMA): X^T = Wl^T x^T, Y^T = Wr^T y^T -> row-major Xh/Yh, bias folded in
    {
      v8f ax0 = {}, ax1 = {}, ay0 = {}, ay1 = {};
      ax0 = wmma_f16(load_a_k16(WlT, m0_0), load_b_k16(xrawh, n0_0), ax0);
      ax1 = wmma_f16(load_a_k16(WlT, m0_1), load_b_k16(xrawh, n0_1), ax1);
      ay0 = wmma_f16(load_a_k16(WrT, m0_0), load_b_k16(yrawh, n0_0), ay0);
      ay1 = wmma_f16(load_a_k16(WrT, m0_1), load_b_k16(yrawh, n0_1), ay1);
      v8h sx0, sx1, sy0, sy1;
#pragma unroll
      for (int j = 0; j < 8; ++j) {
        float bl = bl_s[m0_0 + 8 * hi + j], br = br_s[m0_0 + 8 * hi + j];
        sx0[j] = (_Float16)(ax0[j] + bl);
        sx1[j] = (_Float16)(ax1[j] + bl);
        sy0[j] = (_Float16)(ay0[j] + br);
        sy1[j] = (_Float16)(ay1[j] + br);
      }
      *(v8h*)(Xh + (n0_0 + nn) * 64 + m0_0 + 8 * hi) = sx0;
      *(v8h*)(Xh + (n0_1 + nn) * 64 + m0_1 + 8 * hi) = sx1;
      *(v8h*)(Yh + (n0_0 + nn) * 64 + m0_0 + 8 * hi) = sy0;
      *(v8h*)(Yh + (n0_1 + nn) * 64 + m0_1 + 8 * hi) = sy1;
    }
    __syncthreads();

    // phase 2a (WMMA): xq^T = Wqk X^T (waves 0-3) / yk^T = Wqk Y^T (waves 4-7)
    if (wave < 4) {
      int n0 = wave * 16;
      v8f acc = {};
      acc = wmma_f16(load_a_rm64(Wqkh, 0, 0),  load_b_nk64(Xh, 0,  n0), acc);
      acc = wmma_f16(load_a_rm64(Wqkh, 0, 32), load_b_nk64(Xh, 32, n0), acc);
      v8h s;
#pragma unroll
      for (int j = 0; j < 8; ++j) s[j] = (_Float16)acc[j];
      *(v8h*)(xqh + (n0 + nn) * 16 + 8 * hi) = s;           // row-major [b][d]
    } else {
      int n0 = (wave - 4) * 16;
      v8f acc = {};
      acc = wmma_f16(load_a_rm64(Wqkh, 0, 0),  load_b_nk64(Yh, 0,  n0), acc);
      acc = wmma_f16(load_a_rm64(Wqkh, 0, 32), load_b_nk64(Yh, 32, n0), acc);
      v8h s;
#pragma unroll
      for (int j = 0; j < 8; ++j) s[j] = (_Float16)acc[j];
      *(v8h*)(ykh + (n0 + nn) * 16 + 8 * hi) = s;           // row-major [e][d]
    }
    // phase 2b (WMMA): yv = Y Wv^T + bv, D column-frags stored as yv^T rows (contiguous)
    {
      v8f a0 = {}, a1 = {};
#pragma unroll
      for (int k0 = 0; k0 < 64; k0 += 32) {
        a0 = wmma_f16(load_a_rm64(Yh, m0_0, k0), load_b_nk64(Wvh, k0, n0_0), a0);
        a1 = wmma_f16(load_a_rm64(Yh, m0_1, k0), load_b_nk64(Wvh, k0, n0_1), a1);
      }
      float bv0 = bv_s[n0_0 + nn], bv1 = bv_s[n0_1 + nn];
      v8h s0, s1;
#pragma unroll
      for (int j = 0; j < 8; ++j) {
        s0[j] = (_Float16)(a0[j] + bv0);
        s1[j] = (_Float16)(a1[j] + bv1);
      }
      *(v8h*)(yvT + (n0_0 + nn) * 64 + m0_0 + 8 * hi) = s0;
      *(v8h*)(yvT + (n0_1 + nn) * 64 + m0_1 + 8 * hi) = s1;
    }
    __syncthreads();

    // phase 3 (WMMA): E^T = yk xq^T (K=16 padded) -> row-major E[b][e], contiguous f32 stores
    {
      v8f a0 = {}, a1 = {};
      a0 = wmma_f16(load_a_k16(ykh, m0_0), load_b_k16(xqh, n0_0), a0);
      a1 = wmma_f16(load_a_k16(ykh, m0_1), load_b_k16(xqh, n0_1), a1);
      float* e0 = E + (n0_0 + nn) * 64 + m0_0 + 8 * hi;
      float* e1 = E + (n0_1 + nn) * 64 + m0_1 + 8 * hi;
      v4f p0, p1, p2, p3;
#pragma unroll
      for (int j = 0; j < 4; ++j) { p0[j] = a0[j]; p1[j] = a0[4 + j]; p2[j] = a1[j]; p3[j] = a1[4 + j]; }
      ((v4f*)e0)[0] = p0; ((v4f*)e0)[1] = p1;
      ((v4f*)e1)[0] = p2; ((v4f*)e1)[1] = p3;
    }
    __syncthreads();

    // phase 4: row softmax, 4 threads/row, wave32 shfl_xor reductions; store attn=exp/rowsum
    {
      int r = tid >> 2, q = tid & 3;
      float* erow = E + r * 64 + q * 16;
      v4f e0 = ((const v4f*)erow)[0], e1 = ((const v4f*)erow)[1];
      v4f e2 = ((const v4f*)erow)[2], e3 = ((const v4f*)erow)[3];
      float v[16];
#pragma unroll
      for (int j = 0; j < 4; ++j) { v[j] = e0[j]; v[4 + j] = e1[j]; v[8 + j] = e2[j]; v[12 + j] = e3[j]; }
      float mx = v[0];
#pragma unroll
      for (int j = 1; j < 16; ++j) mx = fmaxf(mx, v[j]);
      mx = fmaxf(mx, __shfl_xor(mx, 1, 32));
      mx = fmaxf(mx, __shfl_xor(mx, 2, 32));
      float s = 0.f;
#pragma unroll
      for (int j = 0; j < 16; ++j) { v[j] = __expf(v[j] - mx); s += v[j]; }
      s += __shfl_xor(s, 1, 32);
      s += __shfl_xor(s, 2, 32);
      float inv = 1.0f / s;
      v4f o0, o1, o2, o3;
#pragma unroll
      for (int j = 0; j < 4; ++j) {
        o0[j] = v[j] * inv; o1[j] = v[4 + j] * inv; o2[j] = v[8 + j] * inv; o3[j] = v[12 + j] * inv;
      }
      ((v4f*)erow)[0] = o0; ((v4f*)erow)[1] = o1; ((v4f*)erow)[2] = o2; ((v4f*)erow)[3] = o3;
    }
    __syncthreads();

    // phase 5: column L1 sums of attn (partials over 16 rows, LDS reduce)
    {
      int e = tid & 63, bq = tid >> 6;
      float s = 0.f;
#pragma unroll
      for (int b = 0; b < 16; ++b) s += E[(bq * 16 + b) * 64 + e];
      red_s[bq * 64 + e] = s;
    }
    __syncthreads();
    if (tid < 64)
      colsum[tid] = 1e-9f + red_s[tid] + red_s[64 + tid] + red_s[128 + tid] + red_s[192 + tid];
    __syncthreads();

    // phase 6: renormalize + in-place f32 -> f16 (read all, barrier, write all; vectorized)
    {
      int base = tid * 16;
      const v4f* ep = (const v4f*)(E + base);
      v4f d0 = ep[0], d1 = ep[1], d2 = ep[2], d3 = ep[3];
      const v4f* cp = (const v4f*)(colsum + (base & 63));
      d0 /= cp[0]; d1 /= cp[1]; d2 /= cp[2]; d3 /= cp[3];
      __syncthreads();
      v8h o0, o1;
#pragma unroll
      for (int j = 0; j < 4; ++j) {
        o0[j] = (_Float16)d0[j]; o0[4 + j] = (_Float16)d1[j];
        o1[j] = (_Float16)d2[j]; o1[4 + j] = (_Float16)d3[j];
      }
      *(v8h*)(attnh + base)     = o0;
      *(v8h*)(attnh + base + 8) = o1;
    }
    __syncthreads();

    // phase 7 (WMMA): yr^T = yv^T attn^T (both operands contiguous);
    // Z = X - yr in-place on Xh: per lane 8 consecutive cols -> v8h RMW
    {
      v8f a0 = {}, a1 = {};
#pragma unroll
      for (int k0 = 0; k0 < 64; k0 += 32) {
        a0 = wmma_f16(load_a_rm64(yvT, m0_0, k0), load_b_nk64(attnh, k0, n0_0), a0);
        a1 = wmma_f16(load_a_rm64(yvT, m0_1, k0), load_b_nk64(attnh, k0, n0_1), a1);
      }
      _Float16* p0 = Xh + (n0_0 + nn) * 64 + m0_0 + 8 * hi;
      _Float16* p1 = Xh + (n0_1 + nn) * 64 + m0_1 + 8 * hi;
      v8h x0 = *(v8h*)p0, x1 = *(v8h*)p1, z0, z1;
#pragma unroll
      for (int j = 0; j < 8; ++j) {
        z0[j] = (_Float16)((float)x0[j] - a0[j]);
        z1[j] = (_Float16)((float)x1[j] - a1[j]);
      }
      *(v8h*)p0 = z0;
      *(v8h*)p1 = z1;
    }
    __syncthreads();

    // phase 8 (WMMA): t^T = Wt Z^T + bt -> each lane stores 8 consecutive floats of t[n][b][:]
    {
      v8f a0 = {}, a1 = {};
#pragma unroll
      for (int k0 = 0; k0 < 64; k0 += 32) {
        a0 = wmma_f16(load_a_rm64(Wth, m0_0, k0), load_b_nk64(Xh, k0, n0_0), a0);
        a1 = wmma_f16(load_a_rm64(Wth, m0_1, k0), load_b_nk64(Xh, k0, n0_1), a1);
      }
      size_t b0 = (size_t)n * 4096 + (n0_0 + nn) * 64 + m0_0 + 8 * hi;
      size_t b1 = (size_t)n * 4096 + (n0_1 + nn) * 64 + m0_1 + 8 * hi;
      v4f q0, q1, q2, q3;
#pragma unroll
      for (int j = 0; j < 8; ++j) {
        float bt = bt_s[m0_0 + 8 * hi + j];
        float v0 = a0[j] + bt;
        float v1 = a1[j] + bt;
        if (j < 4) { q0[j] = v0; q2[j] = v1; } else { q1[j - 4] = v0; q3[j - 4] = v1; }
        bnS[j] += v0 + v1;
        bnQ[j] += v0 * v0 + v1 * v1;
      }
      *(v4f*)(t_out + b0)     = q0;
      *(v4f*)(t_out + b0 + 4) = q1;
      *(v4f*)(t_out + b1)     = q2;
      *(v4f*)(t_out + b1 + 4) = q3;
    }
    __syncthreads();
  }

  // BN partials: reduce across the 16 lanes of each half-wave (same channels), then atomics
#pragma unroll
  for (int j = 0; j < 8; ++j) {
    float s = bnS[j], q = bnQ[j];
#pragma unroll
    for (int m = 1; m < 16; m <<= 1) {
      s += __shfl_xor(s, m, 32);
      q += __shfl_xor(q, m, 32);
    }
    if (nn == 0) {
      int d = m0_0 + 8 * hi + j;
      atomicAdd(&bn_sums[d], s);
      atomicAdd(&bn_sums[64 + d], q);
    }
  }
}

// ---------------- tiny kernels ----------------

__global__ void zero_sums(float* p) { p[threadIdx.x] = 0.0f; }

__global__ void bn_stats(const float* __restrict__ bn_sums, float* __restrict__ stats) {
  int c = threadIdx.x;
  if (c < 64) {
    const float inv = 1.0f / ((float)B_DIM * (float)N_PTS);
    float mean = bn_sums[c] * inv;
    float var  = bn_sums[64 + c] * inv - mean * mean;
    stats[c]      = mean;
    stats[64 + c] = rsqrtf(var + 1e-5f);
  }
}

// ---------------- Pass 3: recompute X, BN+ReLU+residual, max over n ----------------

__global__ __launch_bounds__(256) void finalize_max(
    const float* __restrict__ x, const float* __restrict__ W_l, const float* __restrict__ b_l,
    const float* __restrict__ t_in, const float* __restrict__ stats,
    const float* __restrict__ gamma, const float* __restrict__ beta,
    float* __restrict__ out)
{
  __shared__ float Wl_s[10 * 64];
  __shared__ float bl_s[64], a_s[64], b2_s[64];
  __shared__ float red[4][64];
  const int tid = threadIdx.x;
  const int b = blockIdx.x;
  for (int i = tid; i < 10 * 64; i += 256) Wl_s[i] = W_l[i];
  if (tid < 64) {
    bl_s[tid] = b_l[tid];
    float a = gamma[tid] * stats[64 + tid];       // gamma * rstd
    a_s[tid]  = a;
    b2_s[tid] = beta[tid] - stats[tid] * a;       // beta - mean*gamma*rstd
  }
  __syncthreads();

  const int nl = tid >> 6, c = tid & 63;
  float m = -1e30f;
  for (int n = nl; n < N_PTS; n += 4) {
    const float* xr = &x[((size_t)b * N_PTS + n) * 10];
    float X = bl_s[c];
#pragma unroll
    for (int k = 0; k < 10; ++k) X += xr[k] * Wl_s[k * 64 + c];
    float tv = t_in[(size_t)n * 4096 + b * 64 + c];
    float tn = a_s[c] * tv + b2_s[c];
    m = fmaxf(m, X + fmaxf(tn, 0.0f));
  }
  red[nl][c] = m;
  __syncthreads();
  if (nl == 0) {
    m = fmaxf(fmaxf(red[0][c], red[1][c]), fmaxf(red[2][c], red[3][c]));
    out[b * 64 + c] = m;     // output layout [B, C, 1]
  }
}

// ---------------- host launch ----------------

extern "C" void kernel_launch(void* const* d_in, const int* in_sizes, int n_in,
                              void* d_out, int out_size, void* d_ws, size_t ws_size,
                              hipStream_t stream) {
  const float* x     = (const float*)d_in[0];
  const float* y     = (const float*)d_in[1];
  const float* W_l   = (const float*)d_in[2];
  const float* b_l   = (const float*)d_in[3];
  const float* W_r   = (const float*)d_in[4];
  const float* b_r   = (const float*)d_in[5];
  const float* W_qk  = (const float*)d_in[6];
  const float* W_v   = (const float*)d_in[7];
  const float* b_v   = (const float*)d_in[8];
  const float* W_t   = (const float*)d_in[9];
  const float* b_t   = (const float*)d_in[10];
  const float* gamma = (const float*)d_in[11];
  const float* beta  = (const float*)d_in[12];
  float* out = (float*)d_out;

  float* t_ws = (float*)d_ws;                               // [N][B][C] f32 = 128 MiB
  size_t tElems = (size_t)N_PTS * B_DIM * C_DIM;
  float* bn_sums = t_ws + tElems;                           // 128 floats: sum | sumsq
  float* stats   = bn_sums + 128;                           // 128 floats: mean | rstd

  zero_sums<<<1, 128, 0, stream>>>(bn_sums);
  attn_pass1<<<PASS1_WGS, 256, 0, stream>>>(x, y, W_l, b_l, W_r, b_r,
                                            W_qk, W_v, b_v, W_t, b_t,
                                            t_ws, bn_sums);
  bn_stats<<<1, 64, 0, stream>>>(bn_sums, stats);
  finalize_max<<<B_DIM, 256, 0, stream>>>(x, W_l, b_l, t_ws, stats, gamma, beta, out);
}
